// DotProductAttention_56075093016620
// MI455X (gfx1250) — matrix operand order repaired and checked by reading, not compiled, not run on hardware
//
#include <hip/hip_runtime.h>
#include <hip/hip_bf16.h>

// ---------------------------------------------------------------------------
// Dot-product attention for MI455X (gfx1250), bf16 WMMA with f32 accumulate.
//   score = Q @ V^T ; attn = softmax(score) ; context = attn @ V
// Outputs (flat, in order): context [16,1024,512] f32, attn [16,1024,1024] f32
//
// Strategy: all f32->bf16 conversion happens ONCE in bandwidth-bound prep
// kernels (and at the attn store); the two GEMM kernels' K-loops contain only
// 16B/32B vector loads + v_wmma_f32_16x16x32_bf16, keeping the matrix pipe fed.
// ---------------------------------------------------------------------------

typedef __attribute__((ext_vector_type(16))) __bf16 v16bf;
typedef __attribute__((ext_vector_type(8)))  __bf16 v8bf;
typedef __attribute__((ext_vector_type(8)))  float  v8f;

#define NB   16
#define LQ   1024
#define LKV  1024
#define DIM  512

// f32 -> bf16 round-to-nearest-even (used only in prep / store paths)
__device__ __forceinline__ unsigned short f2bf_u16(float f) {
  unsigned u = __builtin_bit_cast(unsigned, f);
  unsigned r = u + 0x7fffu + ((u >> 16) & 1u);
  return (unsigned short)(r >> 16);
}

// A-fragment (16x32 bf16, MxK). Contiguous k order used here (same as B):
// lane L holds row m = L&15, group g = L>>4;
//   element e in [0,15] -> K = g*16 + e
// Source: row-major bf16 row pointer (already offset to the K-chunk base).
__device__ __forceinline__ v16bf load_fragA_bf(const unsigned short* __restrict__ row, int g) {
  return *(const v16bf*)(row + g * 16);
}

// B-fragment (32x16 bf16, KxN). Contiguous k order used here (same as A):
// lane L holds column n = L&15, group g = L>>4; element e -> K = g*16 + e.
// Source row (a column of B) is contiguous bf16: one 32B vector load.
__device__ __forceinline__ v16bf load_fragB_bf(const unsigned short* __restrict__ row, int g) {
  return *(const v16bf*)(row + g * 16);
}

// ---------------------------------------------------------------------------
// Prep kernel 1: elementwise f32 -> bf16 (one thread per float4 group).
// Used for Qbf and Vbf. Pure streaming, bandwidth bound.
// ---------------------------------------------------------------------------
__global__ void cvt_bf16_kernel(const float* __restrict__ src,
                                unsigned short* __restrict__ dst, int n4) {
  int i = blockIdx.x * blockDim.x + threadIdx.x;
  if (i < n4) {
    float4 f = ((const float4*)src)[i];
    uint2 p;
    p.x = (unsigned)f2bf_u16(f.x) | ((unsigned)f2bf_u16(f.y) << 16);
    p.y = (unsigned)f2bf_u16(f.z) | ((unsigned)f2bf_u16(f.w) << 16);
    ((uint2*)dst)[i] = p;
  }
}

// ---------------------------------------------------------------------------
// Prep kernel 2: V [B,LKV,DIM] f32 -> V^T bf16 [B,DIM,LKV].
// 32x32 LDS tile transpose, block (32,8).
// ---------------------------------------------------------------------------
__global__ void vt_bf16_kernel(const float* __restrict__ v,
                               unsigned short* __restrict__ vt) {
  __shared__ unsigned short tile[32][33];
  const int b  = blockIdx.z;
  const int k0 = blockIdx.x * 32;
  const int d0 = blockIdx.y * 32;
  const int tx = threadIdx.x;      // 0..31
  const int ty = threadIdx.y;      // 0..7
#pragma unroll
  for (int j = 0; j < 32; j += 8) {
    float f = v[((size_t)b * LKV + k0 + ty + j) * DIM + d0 + tx];
    tile[ty + j][tx] = f2bf_u16(f);
  }
  __syncthreads();
#pragma unroll
  for (int j = 0; j < 32; j += 8) {
    vt[((size_t)b * DIM + d0 + ty + j) * LKV + k0 + tx] = tile[tx][ty + j];
  }
}

// ---------------------------------------------------------------------------
// Kernel A: scores + softmax.
// One block = one (batch, 16-row Q tile). 16 waves; wave w owns n-tiles
// [w*64, w*64+64) in 4 chunks of 16. Full 16x1024 score block lives in
// accumulator VGPRs (4 tiles x 8 f32 per lane). Softmax via half-wave
// shuffles + LDS cross-wave reduction. attn written f32 to d_out slice and
// bf16 to workspace (for the context GEMM).
// ---------------------------------------------------------------------------
__global__ void attn_scores_softmax_kernel(const unsigned short* __restrict__ qbf,
                                           const unsigned short* __restrict__ vbf,
                                           float* __restrict__ attn,
                                           unsigned short* __restrict__ attn_bf) {
  __shared__ float red[16][16];   // [wave][row]
  const int lane = threadIdx.x & 31;
  const int w    = threadIdx.x >> 5;   // 0..15
  const int g    = lane >> 4;          // half-wave group
  const int n    = lane & 15;          // column within tile / A-row
  const int b    = blockIdx.x >> 6;
  const int qb   = (blockIdx.x & 63) << 4;

  const unsigned short* qrow = qbf + ((size_t)b * LQ + qb + n) * DIM;  // A rows (m=n)
  const unsigned short* vrow0 =
      vbf + ((size_t)b * LKV + (w * 4) * 16 + n) * DIM;                // B col n tile 0

  v8f acc[4];
#pragma unroll
  for (int t = 0; t < 4; ++t) acc[t] = (v8f)0.0f;

  for (int kc = 0; kc < DIM; kc += 32) {
    v16bf af = load_fragA_bf(qrow + kc, g);
#pragma unroll
    for (int t = 0; t < 4; ++t) {
      v16bf bf = load_fragB_bf(vrow0 + (size_t)t * 16 * DIM + kc, g);
      acc[t] = __builtin_amdgcn_wmma_f32_16x16x32_bf16(
          false, af, false, bf, (short)0, acc[t], false, false);
    }
  }

  // C/D layout: element i of v8f at lane L is (row m = i + 8*g, col n = L&15).
  // ---- row max (over all 1024 columns) ----
  float gmax[8];
  {
    float rmax[8];
#pragma unroll
    for (int i = 0; i < 8; ++i) {
      float m = -3.4e38f;
#pragma unroll
      for (int t = 0; t < 4; ++t) m = fmaxf(m, acc[t][i]);
      for (int d = 1; d < 16; d <<= 1) m = fmaxf(m, __shfl_xor(m, d, 32));
      rmax[i] = m;
    }
    if (n == 0) {
#pragma unroll
      for (int i = 0; i < 8; ++i) red[w][i + 8 * g] = rmax[i];
    }
    __syncthreads();
#pragma unroll
    for (int i = 0; i < 8; ++i) {
      float m = -3.4e38f;
      for (int w2 = 0; w2 < 16; ++w2) m = fmaxf(m, red[w2][i + 8 * g]);
      gmax[i] = m;
    }
    __syncthreads();   // red reused below
  }

  // ---- exp + row sum ----
  float gsum[8];
  {
    float rsum[8];
#pragma unroll
    for (int i = 0; i < 8; ++i) {
      float s = 0.0f;
#pragma unroll
      for (int t = 0; t < 4; ++t) {
        float e = __expf(acc[t][i] - gmax[i]);
        acc[t][i] = e;
        s += e;
      }
      for (int d = 1; d < 16; d <<= 1) s += __shfl_xor(s, d, 32);
      rsum[i] = s;
    }
    if (n == 0) {
#pragma unroll
      for (int i = 0; i < 8; ++i) red[w][i + 8 * g] = rsum[i];
    }
    __syncthreads();
#pragma unroll
    for (int i = 0; i < 8; ++i) {
      float s = 0.0f;
      for (int w2 = 0; w2 < 16; ++w2) s += red[w2][i + 8 * g];
      gsum[i] = s;
    }
  }

  // ---- normalize + store attn (f32 to output, bf16 to workspace) ----
#pragma unroll
  for (int i = 0; i < 8; ++i) {
    const float inv = 1.0f / gsum[i];
    const size_t rbase = ((size_t)b * LQ + qb + i + 8 * g) * LKV;
    float* orow = attn + rbase;
    unsigned short* brow = attn_bf + rbase;
#pragma unroll
    for (int t = 0; t < 4; ++t) {
      const int col = (w * 4 + t) * 16 + n;
      const float p = acc[t][i] * inv;
      orow[col] = p;
      brow[col] = f2bf_u16(p);
    }
  }
}

// ---------------------------------------------------------------------------
// Kernel B: context = attn @ V.
// Block = (batch, 16-row Q tile, 128 d-columns); 8 waves x 16 d each.
// A-frag: bf16 attn from workspace. B-frag: bf16 V^T rows from workspace.
// Inner loop: pure vector loads + WMMA.
// ---------------------------------------------------------------------------
__global__ void attn_context_kernel(const unsigned short* __restrict__ attn_bf,
                                    const unsigned short* __restrict__ vt,
                                    float* __restrict__ out) {
  const int lane  = threadIdx.x & 31;
  const int w     = threadIdx.x >> 5;   // 0..7
  const int g     = lane >> 4;
  const int n     = lane & 15;
  const int b     = blockIdx.x >> 6;
  const int qb    = (blockIdx.x & 63) << 4;
  const int dbase = blockIdx.y * 128 + w * 16;

  const unsigned short* arow = attn_bf + ((size_t)b * LQ + qb + n) * LKV;   // A rows
  const unsigned short* vtrow = vt + ((size_t)b * DIM + dbase + n) * LKV;   // B col n

  v8f acc = (v8f)0.0f;
  for (int kc = 0; kc < LKV; kc += 32) {
    v16bf af = load_fragA_bf(arow + kc, g);
    v16bf bf = load_fragB_bf(vtrow + kc, g);
    acc = __builtin_amdgcn_wmma_f32_16x16x32_bf16(
        false, af, false, bf, (short)0, acc, false, false);
  }

#pragma unroll
  for (int i = 0; i < 8; ++i) {
    out[((size_t)b * LQ + qb + i + 8 * g) * DIM + dbase + n] = acc[i];
  }
}

// ---------------------------------------------------------------------------
extern "C" void kernel_launch(void* const* d_in, const int* in_sizes, int n_in,
                              void* d_out, int out_size, void* d_ws, size_t ws_size,
                              hipStream_t stream) {
  const float* q = (const float*)d_in[0];
  const float* v = (const float*)d_in[1];

  float* ctx  = (float*)d_out;                        // [16,1024,512]
  float* attn = ctx + (size_t)NB * LQ * DIM;          // [16,1024,1024]

  // Workspace layout (bf16 as ushort):
  //   qbf  [NB,LQ,DIM]   16 MB @ 0
  //   vbf  [NB,LKV,DIM]  16 MB @ 16 MB
  //   vbft [NB,DIM,LKV]  16 MB @ 32 MB
  //   pbf  [NB,LQ,LKV]   32 MB @ 48 MB      (bf16 attn)
  const size_t nQ = (size_t)NB * LQ * DIM;    // 8,388,608
  const size_t nV = (size_t)NB * LKV * DIM;   // 8,388,608
  const size_t nP = (size_t)NB * LQ * LKV;    // 16,777,216
  unsigned short* qbf  = (unsigned short*)d_ws;
  unsigned short* vbf  = qbf + nQ;
  unsigned short* vbft = vbf + nV;
  unsigned short* pbf  = vbft + nV;

  // 1) one-time f32 -> bf16 conversions (bandwidth bound)
  {
    const int n4q = (int)(nQ / 4);
    const int n4v = (int)(nV / 4);
    cvt_bf16_kernel<<<(n4q + 255) / 256, 256, 0, stream>>>(q, qbf, n4q);
    cvt_bf16_kernel<<<(n4v + 255) / 256, 256, 0, stream>>>(v, vbf, n4v);
    vt_bf16_kernel<<<dim3(LKV / 32, DIM / 32, NB), dim3(32, 8, 1), 0, stream>>>(v, vbft);
  }

  // 2) scores (Q·V^T, bf16 WMMA) + softmax -> f32 attn output + bf16 attn ws
  attn_scores_softmax_kernel<<<dim3(NB * (LQ / 16)), dim3(512), 0, stream>>>(
      qbf, vbf, attn, pbf);

  // 3) context = attn · V (bf16 WMMA) -> context output slice
  attn_context_kernel<<<dim3(NB * (LQ / 16), DIM / 128), dim3(256), 0, stream>>>(
      pbf, vbft, ctx);
}
